// MidLevel_55972013802038
// MI455X (gfx1250) — matrix operand order repaired and checked by reading, not compiled, not run on hardware
//
#include <hip/hip_runtime.h>

typedef __attribute__((ext_vector_type(8)))  __bf16 v8bf;
typedef __attribute__((ext_vector_type(16))) __bf16 v16bf;
typedef __attribute__((ext_vector_type(8)))  float  v8f;

#define BN_SCALE 0.99999500003749968752f   /* 1/sqrt(1+1e-5) */
#define NBLK 27
#define V34  39304
#define V32  32768
#define V30  27000

__device__ __forceinline__ v8bf bf8_zero() {
    v8bf z;
#pragma unroll
    for (int i = 0; i < 8; ++i) z[i] = (__bf16)0.0f;
    return z;
}
__device__ __forceinline__ v16bf bf16cat(v8bf lo, v8bf hi) {
    return __builtin_shufflevector(lo, hi, 0,1,2,3,4,5,6,7,8,9,10,11,12,13,14,15);
}
__device__ __forceinline__ int iclamp(int x, int lo, int hi) {
    return x < lo ? lo : (x > hi ? hi : x);
}

// ---------------------------------------------------------------------------
// Pack OIDHW f32 weights into WMMA B-fragment order (bf16).
// Fragment layout: [tap][chunk][group][lane][16 halves]; per-lane element i
// holds B(K = (lane>=16?8:0)+(i<8?i:8+i), N = lane%16), outch = g*16+N,
// in-channel = chunk*32 + K (zero if >= CIN).
// ---------------------------------------------------------------------------
template<int CIN, int NG>
__global__ void pack_w3(const float* __restrict__ w, __bf16* __restrict__ dst) {
    constexpr int NCH = (CIN + 31) / 32;
    int idx = blockIdx.x * blockDim.x + threadIdx.x;
    if (idx >= 27 * NCH * NG * 32) return;
    int lane = idx & 31;
    int g    = (idx >> 5) % NG;
    int ch   = (idx / (32 * NG)) % NCH;
    int tap  = idx / (32 * NG * NCH);
    int outch = g * 16 + (lane & 15);
    __bf16* p = dst + (size_t)(((tap * NCH + ch) * NG + g) * 512 + lane * 16);
#pragma unroll
    for (int i = 0; i < 16; ++i) {
        int kb = ((lane >= 16) ? 8 : 0) + (i < 8 ? i : 8 + i);
        int cg = ch * 32 + kb;
        float v = (cg < CIN) ? w[((size_t)outch * CIN + cg) * 27 + tap] : 0.0f;
        p[i] = (__bf16)v;
    }
}

// ---------------------------------------------------------------------------
// ConvTranspose3d 32->32, k=4, s=2, 16^3 -> 34^3, fused relu(bn), bf16 NHWC out
// w layout IODHW: w[ci][oc][kd][kh][kw]
// ---------------------------------------------------------------------------
__global__ void convT_kernel(const float* __restrict__ prev,
                             const float* __restrict__ w,
                             __bf16* __restrict__ out) {
    int idx = blockIdx.x * blockDim.x + threadIdx.x;
    if (idx >= NBLK * V34 * 32) return;
    int och = idx & 31;
    int v   = (idx >> 5) % V34;
    int n   = idx / (32 * V34);
    int od = v / (34*34); int rem = v % (34*34);
    int oh = rem / 34;    int ow  = rem % 34;
    float acc = 0.0f;
    for (int kd = (od & 1); kd < 4; kd += 2) {
        int id = (od - kd) >> 1; if (id < 0 || id >= 16) continue;
        for (int kh = (oh & 1); kh < 4; kh += 2) {
            int ih = (oh - kh) >> 1; if (ih < 0 || ih >= 16) continue;
            for (int kw = (ow & 1); kw < 4; kw += 2) {
                int iw = (ow - kw) >> 1; if (iw < 0 || iw >= 16) continue;
                const float* wp = w + (size_t)((och * 4 + kd) * 4 + kh) * 4 + kw;
                const float* ip = prev + (((size_t)n * 32 * 16 + id) * 16 + ih) * 16 + iw;
#pragma unroll 4
                for (int ci = 0; ci < 32; ++ci)
                    acc += ip[(size_t)ci * 4096] * wp[(size_t)ci * 2048];
            }
        }
    }
    float r = acc * BN_SCALE;
    r = r > 0.0f ? r : 0.0f;
    out[((size_t)n * V34 + v) * 32 + och] = (__bf16)r;
}

// ---------------------------------------------------------------------------
// ts1: replicate-pad(2) conv3 1->16 on the carved tsdf blocks, relu(bn)
// ---------------------------------------------------------------------------
__global__ void ts1_kernel(const float* __restrict__ tsdf,
                           const float* __restrict__ w,
                           __bf16* __restrict__ out) {
    int idx = blockIdx.x * blockDim.x + threadIdx.x;
    if (idx >= NBLK * V34 * 16) return;
    int och = idx & 15;
    int v   = (idx >> 4) % V34;
    int n   = idx / (16 * V34);
    int g0 = n / 9, g1 = (n / 3) % 3, g2 = n % 3;
    int od = v / (34*34); int rem = v % (34*34);
    int oh = rem / 34;    int ow  = rem % 34;
    float acc = 0.0f;
#pragma unroll
    for (int tap = 0; tap < 27; ++tap) {
        int kd = tap / 9, kh = (tap / 3) % 3, kw = tap % 3;
        int b0 = iclamp(od - 2 + kd, 0, 31);
        int b1 = iclamp(oh - 2 + kh, 0, 31);
        int b2 = iclamp(ow - 2 + kw, 0, 31);
        float x = tsdf[((size_t)(g0 * 32 + b0) * 96 + (g1 * 32 + b1)) * 96 + (g2 * 32 + b2)];
        acc += x * w[och * 27 + tap];
    }
    float r = acc * BN_SCALE;
    r = r > 0.0f ? r : 0.0f;
    out[((size_t)n * V34 + v) * 16 + och] = (__bf16)r;
}

// ---------------------------------------------------------------------------
// Generic 3x3x3 conv via bf16 WMMA.
//   - workgroup = 256 threads (8 waves); packed weights staged in LDS once
//   - each wave computes MT=2 tiles of 16 output voxels x NG*16 channels
//   - fully unrolled 27-tap loop; out-of-range rows are CLAMPED to the last
//     voxel (loads stay unconditional & in-bounds; dead rows never stored)
// in0: primary bf16 channel-last tensor (C0 ch/voxel), in1: optional concat
// tensor (C1 ch/voxel, channels C0..C0+C1 of the virtual input).
// OFF = 0 (VALID) or -2 (pad-2), CLAMP = replicate padding.
// ---------------------------------------------------------------------------
template<int SIN, int SOUT, int OFF, bool CLAMP, int C0, int C1, int NG, bool RELU>
__global__ __launch_bounds__(256)
void conv3_wmma(const __bf16* __restrict__ in0, const __bf16* __restrict__ in1,
                const __bf16* __restrict__ wpk, __bf16* __restrict__ out) {
    constexpr int VOUT  = SOUT * SOUT * SOUT;
    constexpr int TILES = (VOUT + 15) / 16;
    constexpr int NCH   = (C1 > 0) ? 2 : 1;
    constexpr int COUT  = NG * 16;
    constexpr int BFRAGS = 27 * NCH * NG;           // 512 halves each
    constexpr int WGTILES = 16;                     // 8 waves x 2 tiles
    constexpr int WGS = (TILES + WGTILES - 1) / WGTILES;

    extern __shared__ char lds_raw[];
    __bf16* sB = (__bf16*)lds_raw;

    // cooperative stage of all B fragments into LDS (16B per copy)
    {
        const v8bf* src = (const v8bf*)wpk;
        v8bf* dst = (v8bf*)sB;
        for (int i = threadIdx.x; i < BFRAGS * 64; i += 256)
            dst[i] = src[i];
    }
    __syncthreads();

    int lane = threadIdx.x & 31;
    int wave = threadIdx.x >> 5;
    int n    = blockIdx.x / WGS;
    int tbase = (blockIdx.x % WGS) * WGTILES + wave * 2;

    int klo = (lane < 16) ? 0 : 8;
    const size_t nbase = (size_t)n * SIN * SIN * SIN;

    // Per-row precomputation: base pointers (VALID) or clamped coords (pad).
    const __bf16* base0_[2];
    const __bf16* base1_[2];
    int zs_[2][3], ys_[2][3], xs_[2][3];
#pragma unroll
    for (int t = 0; t < 2; ++t) {
        int v  = (tbase + t) * 16 + (lane & 15);
        int vc = v < VOUT ? v : (VOUT - 1);          // clamp, never branch
        int od = vc / (SOUT * SOUT);
        int rm = vc % (SOUT * SOUT);
        int oh = rm / SOUT;
        int ow = rm % SOUT;
        if (!CLAMP) {
            size_t vox = nbase + ((size_t)od * SIN + oh) * SIN + ow;  // OFF==0
            base0_[t] = in0 + vox * C0 + klo;
            if (C1 > 0) base1_[t] = in1 + vox * C1 + klo;
        } else {
#pragma unroll
            for (int k = 0; k < 3; ++k) {
                zs_[t][k] = iclamp(od + k + OFF, 0, SIN - 1);
                ys_[t][k] = iclamp(oh + k + OFF, 0, SIN - 1);
                xs_[t][k] = iclamp(ow + k + OFF, 0, SIN - 1);
            }
        }
    }

    v8f acc[2][NG];
#pragma unroll
    for (int t = 0; t < 2; ++t)
#pragma unroll
        for (int g = 0; g < NG; ++g)
            acc[t][g] = {};

#pragma unroll
    for (int tap = 0; tap < 27; ++tap) {
        const int kd = tap / 9, kh = (tap / 3) % 3, kw = tap % 3;

        const __bf16* p0_[2];
        const __bf16* p1_[2];
#pragma unroll
        for (int t = 0; t < 2; ++t) {
            if (!CLAMP) {
                constexpr int stride0 = 1;  // silence unused warnings
                (void)stride0;
                p0_[t] = base0_[t] + (size_t)(((kd * SIN + kh) * SIN + kw) * C0);
                if (C1 > 0) p1_[t] = base1_[t] + (size_t)(((kd * SIN + kh) * SIN + kw) * C1);
            } else {
                size_t vox = nbase + ((size_t)zs_[t][kd] * SIN + ys_[t][kh]) * SIN + xs_[t][kw];
                p0_[t] = in0 + vox * C0 + klo;
                if (C1 > 0) p1_[t] = in1 + vox * C1 + klo;
            }
        }

        // ---- chunk 0: channels [0, C0) of in0 ----
        {
            v16bf a[2];
#pragma unroll
            for (int t = 0; t < 2; ++t) {
                v8bf lo = *(const v8bf*)p0_[t];
                v8bf hi = (C0 == 32) ? *(const v8bf*)(p0_[t] + 16) : bf8_zero();
                a[t] = bf16cat(lo, hi);
            }
#pragma unroll
            for (int g = 0; g < NG; ++g) {
                const __bf16* bp = sB + (size_t)((tap * NCH + 0) * NG + g) * 512 + lane * 16;
                v16bf b = bf16cat(*(const v8bf*)bp, *(const v8bf*)(bp + 8));
#pragma unroll
                for (int t = 0; t < 2; ++t)
                    acc[t][g] = __builtin_amdgcn_wmma_f32_16x16x32_bf16(
                        false, a[t], false, b, (short)0, acc[t][g], false, false);
            }
        }
        // ---- chunk 1: channels [C0, C0+C1) from in1 (C1 <= 16) ----
        if (C1 > 0) {
            v16bf a[2];
#pragma unroll
            for (int t = 0; t < 2; ++t) {
                v8bf lo = *(const v8bf*)p1_[t];
                a[t] = bf16cat(lo, bf8_zero());
            }
#pragma unroll
            for (int g = 0; g < NG; ++g) {
                const __bf16* bp = sB + (size_t)((tap * NCH + 1) * NG + g) * 512 + lane * 16;
                v16bf b = bf16cat(*(const v8bf*)bp, *(const v8bf*)(bp + 8));
#pragma unroll
                for (int t = 0; t < 2; ++t)
                    acc[t][g] = __builtin_amdgcn_wmma_f32_16x16x32_bf16(
                        false, a[t], false, b, (short)0, acc[t][g], false, false);
            }
        }
    }

    // Epilogue: lane holds (M = r + (lane<16?0:8), N = lane&15) for r=0..7
    int mbase = (lane < 16) ? 0 : 8;
#pragma unroll
    for (int t = 0; t < 2; ++t) {
#pragma unroll
        for (int r = 0; r < 8; ++r) {
            int vo = (tbase + t) * 16 + mbase + r;
            if (vo < VOUT) {
                size_t o = ((size_t)n * VOUT + vo) * COUT + (lane & 15);
                float x0 = acc[t][0][r] * BN_SCALE;
                if (RELU) x0 = x0 > 0.0f ? x0 : 0.0f;
                out[o] = (__bf16)x0;
                if (NG == 2) {
                    float x1 = acc[t][NG - 1][r] * BN_SCALE;
                    if (RELU) x1 = x1 > 0.0f ? x1 : 0.0f;
                    out[o + 16] = (__bf16)x1;
                }
            }
        }
    }
}

// ---------------------------------------------------------------------------
// maxpool 3x3x3 stride 3 : 30^3x16 bf16 -> 10^3x16 f32 (channel-last)
// ---------------------------------------------------------------------------
__global__ void maxpool3_kernel(const __bf16* __restrict__ in, float* __restrict__ out) {
    int idx = blockIdx.x * blockDim.x + threadIdx.x;
    if (idx >= NBLK * 1000 * 16) return;
    int c = idx & 15;
    int v = (idx >> 4) % 1000;
    int n = idx / 16000;
    int od = v / 100, oh = (v / 10) % 10, ow = v % 10;
    float m = -3.402823466e+38f;
#pragma unroll
    for (int i = 0; i < 3; ++i)
#pragma unroll
        for (int j = 0; j < 3; ++j)
#pragma unroll
            for (int k = 0; k < 3; ++k) {
                float x = (float)in[((((size_t)n * 30 + od * 3 + i) * 30 + oh * 3 + j) * 30
                                     + ow * 3 + k) * 16 + c];
                m = fmaxf(m, x);
            }
    out[((((size_t)n * 10 + od) * 10 + oh) * 10 + ow) * 16 + c] = m;
}

// conv3 16->8 VALID on 10^3 -> 8^3, relu(bn), f32 channel-last
__global__ void convpb_kernel(const float* __restrict__ in, const float* __restrict__ w,
                              float* __restrict__ out) {
    int idx = blockIdx.x * blockDim.x + threadIdx.x;
    if (idx >= NBLK * 512 * 8) return;
    int oc = idx & 7;
    int v  = (idx >> 3) % 512;
    int n  = idx / 4096;
    int od = v / 64, oh = (v / 8) % 8, ow = v % 8;
    float acc = 0.0f;
    for (int tap = 0; tap < 27; ++tap) {
        int kd = tap / 9, kh = (tap / 3) % 3, kw = tap % 3;
        const float* ip = in + ((((size_t)n * 10 + od + kd) * 10 + oh + kh) * 10 + ow + kw) * 16;
#pragma unroll
        for (int ci = 0; ci < 16; ++ci)
            acc += ip[ci] * w[((size_t)oc * 16 + ci) * 27 + tap];
    }
    float r = acc * BN_SCALE;
    r = r > 0.0f ? r : 0.0f;
    out[((size_t)n * 512 + v) * 8 + oc] = r;
}

// maxpool 4x4x4 stride 4 : 8^3x8 -> 2^3x8 f32
__global__ void maxpool4_kernel(const float* __restrict__ in, float* __restrict__ out) {
    int idx = blockIdx.x * blockDim.x + threadIdx.x;
    if (idx >= NBLK * 8 * 8) return;
    int oc = idx & 7;
    int v  = (idx >> 3) % 8;
    int n  = idx / 64;
    int od = v / 4, oh = (v / 2) % 2, ow = v % 2;
    float m = -3.402823466e+38f;
#pragma unroll
    for (int i = 0; i < 4; ++i)
#pragma unroll
        for (int j = 0; j < 4; ++j)
#pragma unroll
            for (int k = 0; k < 4; ++k) {
                float x = in[((((size_t)n * 8 + od * 4 + i) * 8 + oh * 4 + j) * 8
                              + ow * 4 + k) * 8 + oc];
                m = fmaxf(m, x);
            }
    out[((((size_t)n * 2 + od) * 2 + oh) * 2 + ow) * 8 + oc] = m;
}

// linear 64->3 + softmax + threshold (reference flattens NCDHW order)
__global__ void final_kernel(const float* __restrict__ p4, const float* __restrict__ wl,
                             const float* __restrict__ bl, float* __restrict__ outp) {
    int n = threadIdx.x;
    if (n >= NBLK) return;
    float logit[3];
#pragma unroll
    for (int j = 0; j < 3; ++j) {
        float s = bl[j];
        for (int c = 0; c < 8; ++c)
#pragma unroll
            for (int d = 0; d < 2; ++d)
#pragma unroll
                for (int h = 0; h < 2; ++h)
#pragma unroll
                    for (int w = 0; w < 2; ++w) {
                        float x = p4[((((size_t)n * 2 + d) * 2 + h) * 2 + w) * 8 + c];
                        s += x * wl[j * 64 + ((c * 2 + d) * 2 + h) * 2 + w];
                    }
        logit[j] = s;
    }
    float mx = fmaxf(logit[0], fmaxf(logit[1], logit[2]));
    float e0 = expf(logit[0] - mx), e1 = expf(logit[1] - mx), e2 = expf(logit[2] - mx);
    float inv = 1.0f / (e0 + e1 + e2);
    float p0 = e0 * inv, p1 = e1 * inv, p2 = e2 * inv;
    outp[n * 3 + 0] = p0;
    outp[n * 3 + 1] = p1;
    outp[n * 3 + 2] = p2;
    outp[81 + n] = (p2 > 0.1f) ? 1.0f : 0.0f;
}

// ---------------------------------------------------------------------------
extern "C" void kernel_launch(void* const* d_in, const int* in_sizes, int n_in,
                              void* d_out, int out_size, void* d_ws, size_t ws_size,
                              hipStream_t stream) {
    (void)in_sizes; (void)n_in; (void)out_size; (void)ws_size;
    const float* tsdf   = (const float*)d_in[0];
    const float* prev   = (const float*)d_in[1];
    const float* w_up_t = (const float*)d_in[2];
    const float* w_up_c = (const float*)d_in[3];
    const float* w_ts1  = (const float*)d_in[4];
    const float* w_ts2  = (const float*)d_in[5];
    const float* w_b1a  = (const float*)d_in[6];
    const float* w_b1b  = (const float*)d_in[7];
    const float* w_pa   = (const float*)d_in[8];
    const float* w_pb   = (const float*)d_in[9];
    const float* w_lin  = (const float*)d_in[10];
    const float* b_lin  = (const float*)d_in[11];
    float* outp = (float*)d_out;

    char* ws = (char*)d_ws;
    size_t off = 0;
    auto alloc = [&](size_t bytes) { size_t o = off; off += (bytes + 255) & ~(size_t)255; return o; };
    size_t oA = alloc((size_t)NBLK * V34 * 32 * 2);   // h1, later g1
    size_t oB = alloc((size_t)NBLK * V34 * 16 * 2);   // h2, later p1
    size_t oC = alloc((size_t)NBLK * V32 * 32 * 2);   // prevs, later feat
    size_t oD = alloc((size_t)NBLK * V32 * 16 * 2);   // tfeat, later p2/p3/p4
    size_t oPKupc = alloc(54 * 1024);
    size_t oPKts2 = alloc(27 * 1024);
    size_t oPKb1a = alloc(108 * 1024);
    size_t oPKb1b = alloc(54 * 1024);
    size_t oPKpa  = alloc(27 * 1024);

    __bf16* h1    = (__bf16*)(ws + oA);
    __bf16* g1    = h1;                       // reuse: h1 dead after upc conv
    __bf16* h2    = (__bf16*)(ws + oB);
    __bf16* p1    = h2;                       // reuse: h2 dead after ts2 conv
    __bf16* prevs = (__bf16*)(ws + oC);
    __bf16* feat  = prevs;                    // reuse: prevs dead after b1a
    __bf16* tfeat = (__bf16*)(ws + oD);
    float*  p2    = (float*)(ws + oD);        // reuse: tfeat dead after b1a
    float*  p3    = (float*)(ws + oD + 1728000);
    float*  p4    = (float*)(ws + oD + 1728000 + 442368);
    __bf16* pk_upc = (__bf16*)(ws + oPKupc);
    __bf16* pk_ts2 = (__bf16*)(ws + oPKts2);
    __bf16* pk_b1a = (__bf16*)(ws + oPKb1a);
    __bf16* pk_b1b = (__bf16*)(ws + oPKb1b);
    __bf16* pk_pa  = (__bf16*)(ws + oPKpa);

    // --- pack weights into WMMA B-fragment layout ---
    pack_w3<32, 2><<<7, 256, 0, stream>>>(w_up_c, pk_upc);   // 1728 threads
    pack_w3<16, 1><<<4, 256, 0, stream>>>(w_ts2, pk_ts2);    // 864
    pack_w3<48, 2><<<14, 256, 0, stream>>>(w_b1a, pk_b1a);   // 3456
    pack_w3<32, 2><<<7, 256, 0, stream>>>(w_b1b, pk_b1b);
    pack_w3<32, 1><<<4, 256, 0, stream>>>(w_pa, pk_pa);

    // --- stage 1: h1 = relu(bn(convT(prev))) : [27][34^3][32] bf16 ---
    {
        int total = NBLK * V34 * 32;
        convT_kernel<<<(total + 255) / 256, 256, 0, stream>>>(prev, w_up_t, h1);
    }
    // --- stage 2: h2 = relu(bn(conv_pad2(tsdf blocks, w_ts1))) : [27][34^3][16] ---
    {
        int total = NBLK * V34 * 16;
        ts1_kernel<<<(total + 255) / 256, 256, 0, stream>>>(tsdf, w_ts1, h2);
    }
    // --- stage 3: prevs = bn(conv(h1, w_up_c)) : 34 -> 32, 32ch ---
    // TILES=2048 -> 128 WGs per block; LDS = 27*1*2 KB = 54KB
    conv3_wmma<34, 32, 0, false, 32, 0, 2, false>
        <<<NBLK * 128, 256, 54 * 1024, stream>>>(h1, (const __bf16*)nullptr, pk_upc, prevs);
    // --- stage 4: tfeat = bn(conv(h2, w_ts2)) : 34 -> 32, 16ch ; LDS = 27KB ---
    conv3_wmma<34, 32, 0, false, 16, 0, 1, false>
        <<<NBLK * 128, 256, 27 * 1024, stream>>>(h2, (const __bf16*)nullptr, pk_ts2, tfeat);
    // --- stage 5: g1 = relu(bn(conv_pad2(concat(prevs,tfeat), w_b1a))) : 32 -> 34 ---
    // TILES=2457 -> 154 WGs; LDS = 27*2*2 KB = 108KB
    conv3_wmma<32, 34, -2, true, 32, 16, 2, true>
        <<<NBLK * 154, 256, 108 * 1024, stream>>>(prevs, tfeat, pk_b1a, g1);
    // --- stage 6: feat = bn(conv(g1, w_b1b)) : 34 -> 32, 32ch ; LDS = 54KB ---
    conv3_wmma<34, 32, 0, false, 32, 0, 2, false>
        <<<NBLK * 128, 256, 54 * 1024, stream>>>(g1, (const __bf16*)nullptr, pk_b1b, feat);
    // --- stage 7: p1 = relu(bn(conv(feat, w_pa))) : 32 -> 30, 16ch ---
    // TILES=1688 -> 106 WGs; LDS = 27KB
    conv3_wmma<32, 30, 0, false, 32, 0, 1, true>
        <<<NBLK * 106, 256, 27 * 1024, stream>>>(feat, (const __bf16*)nullptr, pk_pa, p1);
    // --- stage 8: maxpool 3 ---
    {
        int total = NBLK * 1000 * 16;
        maxpool3_kernel<<<(total + 255) / 256, 256, 0, stream>>>(p1, p2);
    }
    // --- stage 9: p3 = relu(bn(conv(p2, w_pb))) : 10 -> 8, 8ch ---
    {
        int total = NBLK * 512 * 8;
        convpb_kernel<<<(total + 255) / 256, 256, 0, stream>>>(p2, w_pb, p3);
    }
    // --- stage 10: maxpool 4 ---
    {
        int total = NBLK * 8 * 8;
        maxpool4_kernel<<<(total + 255) / 256, 256, 0, stream>>>(p3, p4);
    }
    // --- stage 11: linear + softmax + threshold ---
    final_kernel<<<1, 32, 0, stream>>>(p4, w_lin, b_lin, outp);
}
